// Attention_37280316129367
// MI455X (gfx1250) — matrix operand order repaired and checked by reading, not compiled, not run on hardware
//
#include <hip/hip_runtime.h>

// ---------------------------------------------------------------------------
// Bahdanau attention, fused flash-style, BF16 WMMA on gfx1250 (wave32).
// B=32, S=2048, H=1024, 2H=2048.
// ---------------------------------------------------------------------------

typedef __attribute__((ext_vector_type(4)))  __bf16 v4bf;
typedef __attribute__((ext_vector_type(8)))  __bf16 v8bf;
typedef __attribute__((ext_vector_type(16))) __bf16 v16bf;
typedef __attribute__((ext_vector_type(8)))  float  v8f;

#define BB   32
#define SS   2048
#define HH   1024
#define TWOH 2048
#define TM   64                 // enc rows (s) per block
#define CHUNKS (SS / TM)        // 32 chunks per batch
#define APITCH (TWOH + 8)       // bf16 elems per LDS row (+16B pad: bank spread)
#define KT   (TWOH / 32)        // 64 k-tiles
#define NTHREADS 512            // 16 waves: 2 m-groups x 8 n-groups

// ---------------------------------------------------------------------------
// Kernel 0a: retile W_enc [H,2H] fp32 -> bf16 in WMMA B-fragment order.
// B-fragment order used here (contiguous k, matches A): lane l holds column n = l&15,
// K rows kbase..kbase+15 (kbase = 0 for lanes 0-15, 16 for lanes 16-31).
// Frag(nt,kt) occupies 512 contiguous bf16, lane-major -> coalesced loads.
// ---------------------------------------------------------------------------
__global__ void tile_wenc_kernel(const float* __restrict__ W, __bf16* __restrict__ Wt) {
    int gid  = blockIdx.x * 256 + threadIdx.x;    // 0 .. 64*64*32-1
    int lane = gid & 31;
    int kt   = (gid >> 5) & 63;
    int nt   = gid >> 11;
    int n = (nt << 4) + (lane & 15);
    int k = (kt << 5) + (lane & 16);              // lanes 16-31: +16
    const float* src = W + (size_t)n * TWOH + k;
    __bf16* dst = Wt + (size_t)gid * 16;
#pragma unroll
    for (int j = 0; j < 16; ++j) dst[j] = (__bf16)src[j];
}

// ---------------------------------------------------------------------------
// Kernel 0b: dec_proj[b,h] = sum_d dec_state[b,d] * W_dec[h,d]  (tiny GEMV)
// ---------------------------------------------------------------------------
__global__ void dec_proj_kernel(const float* __restrict__ dec,
                                const float* __restrict__ Wd,
                                float* __restrict__ out) {
    __shared__ float dv[HH];
    int b = blockIdx.x >> 2;
    int h = ((blockIdx.x & 3) << 8) + threadIdx.x;
    for (int i = threadIdx.x; i < HH; i += 256) dv[i] = dec[b * HH + i];
    __syncthreads();
    const float4* wr = (const float4*)(Wd + (size_t)h * HH);
    float acc = 0.f;
#pragma unroll 4
    for (int i = 0; i < HH / 4; ++i) {
        float4 w = wr[i];
        acc += w.x * dv[4*i] + w.y * dv[4*i+1] + w.z * dv[4*i+2] + w.w * dv[4*i+3];
    }
    out[b * HH + h] = acc;
}

// ---------------------------------------------------------------------------
// Main kernel: per (b, s-chunk of 64 rows), 512 threads = 16 waves (2m x 8n)
//  - stage enc rows as bf16 into LDS (single HBM pass over enc_state)
//  - WMMA bf16 GEMM: wave tile 32x64 (2 A-frags x 4 B-frags -> 8 wmma/k-step)
//  - epilogue: tanh(dec+x)*W_energy reduced into per-row energies
//  - chunk softmax stats (m,l) + exp-weighted partial context from LDS tile
// ---------------------------------------------------------------------------
__global__ void attn_main_kernel(const float*  __restrict__ enc,
                                 const int*    __restrict__ srclen,
                                 const __bf16* __restrict__ Wt,
                                 const float*  __restrict__ decp,
                                 const float*  __restrict__ Wen,
                                 float* __restrict__ energies,
                                 float* __restrict__ pm,
                                 float* __restrict__ pl,
                                 float* __restrict__ pctx) {
    extern __shared__ char smem[];
    float*  dw    = (float*)smem;          // 1024: dec_proj[b,:]
    float*  we    = dw + HH;               // 1024: W_energy
    float*  e_lds = we + HH;               // 64:   per-row energies
    float*  w_lds = e_lds + TM;            // 64:   per-row softmax weights
    __bf16* A_lds = (__bf16*)(w_lds + TM); // 64 x APITCH bf16 (offset 8704B, 16B aligned)

    const int tid   = threadIdx.x;
    const int b     = blockIdx.x / CHUNKS;
    const int chunk = blockIdx.x % CHUNKS;
    const int s0    = chunk * TM;

    for (int i = tid; i < HH; i += NTHREADS) { dw[i] = decp[b * HH + i]; we[i] = Wen[i]; }
    if (tid < TM) e_lds[tid] = 0.f;

    // ---- stage A: 64 rows x 2048 fp32 -> bf16 LDS (coalesced float4) ----
    const float* Ab = enc + ((size_t)b * SS + s0) * TWOH;
    for (int i = tid; i < TM * (TWOH / 4); i += NTHREADS) {
        int r = i >> 9, c = i & 511;
        float4 v = ((const float4*)(Ab + (size_t)r * TWOH))[c];
        v4bf p = { (__bf16)v.x, (__bf16)v.y, (__bf16)v.z, (__bf16)v.w };
        *(v4bf*)(A_lds + (size_t)r * APITCH + 4 * c) = p;
    }
    __syncthreads();

    const int wave = tid >> 5, lane = tid & 31;
    const int mw = wave & 1;        // 2 m-groups of 32 rows
    const int nw = wave >> 1;       // 8 n-groups of 64 cols
    const int l15 = lane & 15, hi = lane >> 4;

    // ---- GEMM + energy epilogue over 2 N-chunks of 512 cols ----
    for (int nc = 0; nc < 2; ++nc) {
        const int ncol0 = nc * 512 + nw * 64;
        v8f acc[2][4];
#pragma unroll
        for (int am = 0; am < 2; ++am)
#pragma unroll
            for (int t = 0; t < 4; ++t)
                acc[am][t] = (v8f){0.f, 0.f, 0.f, 0.f, 0.f, 0.f, 0.f, 0.f};

        for (int k0 = 0; k0 < TWOH; k0 += 32) {
            // two A fragments 16x32: rows mw*32 + am*16 + l15,
            // lane holds K {lo..lo+15}, lo = hi*16
            v16bf af[2];
#pragma unroll
            for (int am = 0; am < 2; ++am) {
                const int arow = mw * 32 + am * 16 + l15;
                const __bf16* pa = A_lds + (size_t)arow * APITCH + k0 + hi * 16;
                v8bf a0 = *(const v8bf*)pa;
                v8bf a1 = *(const v8bf*)(pa + 8);
                af[am] = __builtin_shufflevector(a0, a1,
                    0,1,2,3,4,5,6,7,8,9,10,11,12,13,14,15);
            }

            const size_t fbase = ((size_t)(ncol0 >> 4) * KT + (k0 >> 5)) * 512
                               + (size_t)lane * 16;
            __builtin_prefetch(Wt + fbase + 512, 0, 3);   // next k-tile
#pragma unroll
            for (int t = 0; t < 4; ++t) {
                const __bf16* pb = Wt + fbase + (size_t)t * (KT * 512);
                v8bf b0 = *(const v8bf*)pb;
                v8bf b1 = *(const v8bf*)(pb + 8);
                v16bf bf = __builtin_shufflevector(b0, b1,
                    0,1,2,3,4,5,6,7,8,9,10,11,12,13,14,15);
#pragma unroll
                for (int am = 0; am < 2; ++am)
                    acc[am][t] = __builtin_amdgcn_wmma_f32_16x16x32_bf16(
                        false, af[am], false, bf, (short)0, acc[am][t], false, false);
            }
        }

        // epilogue: energies += tanh(dec + x) * W_energy over this wave's cols
        float dwv[4], wev[4];
#pragma unroll
        for (int t = 0; t < 4; ++t) {
            int h = ncol0 + t * 16 + l15;
            dwv[t] = dw[h]; wev[t] = we[h];
        }
#pragma unroll
        for (int am = 0; am < 2; ++am) {
            const int rbase = mw * 32 + am * 16 + 8 * hi;
#pragma unroll
            for (int r = 0; r < 8; ++r) {
                float er = 0.f;
#pragma unroll
                for (int t = 0; t < 4; ++t)
                    er += tanhf(dwv[t] + acc[am][t][r]) * wev[t];
                atomicAdd(&e_lds[rbase + r], er);
            }
        }
    }
    __syncthreads();

    // ---- chunk-local softmax stats ----
    const int len = srclen[b];
    if (tid < TM) {
        int s = s0 + tid;
        energies[(size_t)b * SS + s] = (s < len) ? e_lds[tid] : -INFINITY;
    }
    if (tid == 0) {
        float m = -INFINITY;
        for (int r = 0; r < TM; ++r)
            if (s0 + r < len) m = fmaxf(m, e_lds[r]);
        float l = 0.f;
        for (int r = 0; r < TM; ++r) {
            float w = (s0 + r < len) ? __expf(e_lds[r] - m) : 0.f;
            w_lds[r] = w; l += w;
        }
        pm[b * CHUNKS + chunk] = m;
        pl[b * CHUNKS + chunk] = l;
    }
    __syncthreads();

    // ---- partial context from the LDS tile (no second HBM pass) ----
    float* dst = pctx + ((size_t)b * CHUNKS + chunk) * TWOH;
    const int c0 = tid * 4;                 // 512 threads x 4 cols = 2048
    float cacc[4];
#pragma unroll
    for (int j = 0; j < 4; ++j) cacc[j] = 0.f;
    for (int r = 0; r < TM; ++r) {
        float wgt = w_lds[r];
        if (wgt != 0.f) {
            v4bf row = *(const v4bf*)(A_lds + (size_t)r * APITCH + c0);
#pragma unroll
            for (int j = 0; j < 4; ++j) cacc[j] += wgt * (float)row[j];
        }
    }
#pragma unroll
    for (int j = 0; j < 4; ++j) dst[c0 + j] = cacc[j];
}

// ---------------------------------------------------------------------------
// Finalize: per batch merge 32 chunk partials; write alphas then context.
// d_out = [context: B*2H][alphas: B*S]
// ---------------------------------------------------------------------------
__global__ void attn_finalize_kernel(const float* __restrict__ energies,
                                     const float* __restrict__ pm,
                                     const float* __restrict__ pl,
                                     const float* __restrict__ pctx,
                                     const int*   __restrict__ srclen,
                                     float* __restrict__ out) {
    __shared__ float sscale[CHUNKS];
    __shared__ float gstat[2];   // [0]=m, [1]=L
    const int b = blockIdx.x, tid = threadIdx.x;
    if (tid == 0) {
        float m = -INFINITY;
        for (int i = 0; i < CHUNKS; ++i) m = fmaxf(m, pm[b * CHUNKS + i]);
        float L = 0.f;
        for (int i = 0; i < CHUNKS; ++i) {
            float li = pl[b * CHUNKS + i];
            float sc = (li > 0.f) ? __expf(pm[b * CHUNKS + i] - m) : 0.f;
            sscale[i] = sc;
            L += sc * li;
        }
        float inv = 1.f / L;
        for (int i = 0; i < CHUNKS; ++i) sscale[i] *= inv;
        gstat[0] = m; gstat[1] = L;
    }
    __syncthreads();
    const float m = gstat[0], invL = 1.f / gstat[1];
    const int len = srclen[b];
    // alphas
    for (int s = tid; s < SS; s += 256) {
        float e = energies[(size_t)b * SS + s];
        out[(size_t)BB * TWOH + (size_t)b * SS + s] = (s < len) ? __expf(e - m) * invL : 0.f;
    }
    // context
    for (int c = tid; c < TWOH; c += 256) {
        float acc = 0.f;
        for (int i = 0; i < CHUNKS; ++i)
            acc += sscale[i] * pctx[((size_t)b * CHUNKS + i) * TWOH + c];
        out[(size_t)b * TWOH + c] = acc;
    }
}

// ---------------------------------------------------------------------------
extern "C" void kernel_launch(void* const* d_in, const int* in_sizes, int n_in,
                              void* d_out, int out_size, void* d_ws, size_t ws_size,
                              hipStream_t stream) {
    const float* dec_state  = (const float*)d_in[0];
    const float* enc_state  = (const float*)d_in[1];
    const int*   src_length = (const int*)  d_in[2];
    const float* W_enc      = (const float*)d_in[3];
    const float* W_dec      = (const float*)d_in[4];
    const float* W_energy   = (const float*)d_in[5];
    float* out = (float*)d_out;

    char* ws = (char*)d_ws;
    __bf16* Wt   = (__bf16*)(ws);                       // 4,194,304 B
    float*  decp = (float*) (ws + 4194304);             //   131,072 B
    float*  energ= (float*) (ws + 4325376);             //   262,144 B
    float*  pm   = (float*) (ws + 4587520);             //     4,096 B
    float*  pl   = (float*) (ws + 4591616);             //     4,096 B
    float*  pctx = (float*) (ws + 4595712);             // 8,388,608 B

    tile_wenc_kernel<<<512, 256, 0, stream>>>(W_enc, Wt);
    dec_proj_kernel<<<128, 256, 0, stream>>>(dec_state, W_dec, decp);

    const size_t smem = (size_t)(HH + HH + TM + TM) * sizeof(float)
                      + (size_t)TM * APITCH * sizeof(__bf16);   // 271,872 B < 320 KB
    attn_main_kernel<<<BB * CHUNKS, NTHREADS, smem, stream>>>(
        enc_state, src_length, Wt, decp, W_energy, energ, pm, pl, pctx);

    attn_finalize_kernel<<<BB, 256, 0, stream>>>(energ, pm, pl, pctx, src_length, out);
}
